// Attention_27951647162870
// MI455X (gfx1250) — compile-verified, matrix operand order checked by reading, not run on hardware
//
#include <hip/hip_runtime.h>
#include <hip/hip_bf16.h>

typedef __attribute__((ext_vector_type(16))) _Float16     v16h;
typedef __attribute__((ext_vector_type(8)))  _Float16     v8h;
typedef __attribute__((ext_vector_type(8)))  float        v8f;
typedef __attribute__((ext_vector_type(4)))  unsigned int u32x4;
typedef __attribute__((ext_vector_type(8)))  unsigned int u32x8;

// ---------------------------------------------------------------------------
// Fragment loader for V_WMMA_F32_16X16X32_F16.
// 16-bit A (16x32, MxK): lane L -> row = L&15; elems 0..7 = K=(L>>4)*8..+7,
// elems 8..15 = K=16+(L>>4)*8..+7.  B (32x16) uses the same pattern with the
// lane indexing column N (column n of B == row n of W for A @ W^T).
// Works for global or LDS-backed pointers (lowered to b128 loads).
// ---------------------------------------------------------------------------
__device__ __forceinline__ v16h load_frag(const _Float16* base,
                                          int stride, int row, int k0, int half) {
  const _Float16* r = base + (size_t)row * stride + k0 + half * 8;
  v8h lo = *(const v8h*)(r);
  v8h hi = *(const v8h*)(r + 16);
  return __builtin_shufflevector(lo, hi, 0,1,2,3,4,5,6,7,8,9,10,11,12,13,14,15);
}

// ---------------------------------------------------------------------------
// TDM: DMA a 2-D f16 tile [rows x 32] (dim0 contiguous, length 32, stride K)
// from global into LDS.  D# per CDNA5 ISA.
// Issued by one wave; completion tracked with TENSORcnt.
// ---------------------------------------------------------------------------
__device__ __forceinline__ void tdm_load_tile_f16(const _Float16* gsrc,
                                                  unsigned ldsOff,
                                                  int K, int N, int rows) {
  unsigned long long ga = (unsigned long long)(size_t)gsrc;
  u32x4 g0;
  g0[0] = 1u;                                   // count=1, user descriptor
  g0[1] = ldsOff;                               // lds_addr
  g0[2] = (unsigned)ga;                         // global_addr[31:0]
  g0[3] = (unsigned)((ga >> 32) & 0x01FFFFFFu)  // global_addr[56:32]
          | (2u << 30);                         // type=2 ("image")
  u32x8 g1;
  g1[0] = 1u << 16;                             // data_size=1 (2 bytes)
  g1[1] = ((unsigned)K & 0xFFFFu) << 16;        // tensor_dim0[15:0]
  g1[2] = ((unsigned)K >> 16) | (((unsigned)N & 0xFFFFu) << 16); // dim0 hi, dim1 lo
  g1[3] = ((unsigned)N >> 16) | (32u << 16);    // dim1 hi, tile_dim0=32
  g1[4] = (unsigned)rows;                       // tile_dim1, tile_dim2=0
  g1[5] = (unsigned)K;                          // tensor_dim0_stride[31:0]
  g1[6] = 0u;                                   // stride0 hi, stride1 lo
  g1[7] = 0u;                                   // stride1 hi
  asm volatile("tensor_load_to_lds %0, %1" :: "s"(g0), "s"(g1) : "memory");
}

// ---------------------------------------------------------------------------
// f32 -> f16 streaming convert
// ---------------------------------------------------------------------------
__global__ void cvt_kernel(const float* __restrict__ in, _Float16* __restrict__ out,
                           size_t n) {
  size_t i = (size_t)blockIdx.x * blockDim.x + threadIdx.x;
  if (i < n) out[i] = (_Float16)in[i];
}

// ---------------------------------------------------------------------------
// C[M,N] (f32) = A[M,K] (f16) @ W[N,K]^T (f16), WMMA f32 accumulate.
// 256 threads = 8 waves, wave -> 16x64 strip, block -> 128x64 tile.
// B tile (64 rows x 32 K) staged by the Tensor Data Mover into double-buffered
// LDS; TDM for step k+1 overlaps the WMMAs of step k.  A fragments are
// double-buffered in registers.  All 4 B gathers issue before the WMMA chain
// so DS latency overlaps the first matrix ops (partial dscnt waits).
// ---------------------------------------------------------------------------
__global__ void gemm_f16_wmma(const _Float16* __restrict__ A,
                              const _Float16* __restrict__ W,
                              float* __restrict__ C,
                              int M, int N, int K) {
  __shared__ __align__(16) _Float16 wtile[2][64][32];   // 2 x 4 KB

  const int lane = threadIdx.x & 31;
  const int w    = threadIdx.x >> 5;
  const int half = lane >> 4;
  const int lid  = lane & 15;
  const int m0 = blockIdx.x * 128 + w * 16;
  const int n0 = blockIdx.y * 64;

  v8f acc[4];
#pragma unroll
  for (int nt = 0; nt < 4; ++nt)
#pragma unroll
    for (int j = 0; j < 8; ++j) acc[nt][j] = 0.0f;

  const _Float16* Wt = W + (size_t)n0 * K;  // tile row origin
  const unsigned ldsBase0 = (unsigned)(size_t)&wtile[0][0][0];
  const unsigned ldsBase1 = (unsigned)(size_t)&wtile[1][0][0];

  // prologue: DMA first B tile, prime first A fragment
  if (w == 0) {
    tdm_load_tile_f16(Wt, ldsBase0, K, N, 64);
    __builtin_amdgcn_s_wait_tensorcnt(0);
  }
  __syncthreads();
  v16h a_cur = load_frag(A, K, m0 + lid, 0, half);

  int buf = 0;
  for (int k0 = 0; k0 < K; k0 += 32) {
    const bool more = (k0 + 32) < K;
    if (more && w == 0)  // async DMA of next B tile into the other buffer
      tdm_load_tile_f16(Wt + (k0 + 32), buf ? ldsBase0 : ldsBase1, K, N, 64);
    v16h a_nxt = a_cur;
    if (more) a_nxt = load_frag(A, K, m0 + lid, k0 + 32, half);

    // batch the 4 B gathers, then the 4 WMMAs (partial dscnt waits)
    const _Float16* bt = &wtile[buf][0][0];
    v16h b0 = load_frag(bt, 32,  0 + lid, 0, half);
    v16h b1 = load_frag(bt, 32, 16 + lid, 0, half);
    v16h b2 = load_frag(bt, 32, 32 + lid, 0, half);
    v16h b3 = load_frag(bt, 32, 48 + lid, 0, half);
    acc[0] = __builtin_amdgcn_wmma_f32_16x16x32_f16(false, a_cur, false, b0,
                                                    (short)0, acc[0], false, false);
    acc[1] = __builtin_amdgcn_wmma_f32_16x16x32_f16(false, a_cur, false, b1,
                                                    (short)0, acc[1], false, false);
    acc[2] = __builtin_amdgcn_wmma_f32_16x16x32_f16(false, a_cur, false, b2,
                                                    (short)0, acc[2], false, false);
    acc[3] = __builtin_amdgcn_wmma_f32_16x16x32_f16(false, a_cur, false, b3,
                                                    (short)0, acc[3], false, false);

    if (more && w == 0) __builtin_amdgcn_s_wait_tensorcnt(0);
    __syncthreads();
    a_cur = a_nxt;
    buf ^= 1;
  }

  // C/D layout: VGPR j holds row (j + 8*half), lane lid holds column lid.
#pragma unroll
  for (int nt = 0; nt < 4; ++nt)
#pragma unroll
    for (int j = 0; j < 8; ++j)
      C[(size_t)(m0 + half * 8 + j) * N + n0 + nt * 16 + lid] = acc[nt][j];
}

// ---------------------------------------------------------------------------
// RoPE + L2-norm + scale + [B,T,NH,64] -> [B,NH,T,64] f16 transpose.
// One wave per (b,t,head); lane i owns the (2i, 2i+1) rotary pair.
// ---------------------------------------------------------------------------
__global__ void rope_norm_kernel(const float* __restrict__ X,
                                 const float* __restrict__ fcos,
                                 const float* __restrict__ fsin,
                                 const float* __restrict__ scales,
                                 _Float16* __restrict__ Out,
                                 int Bc, int Tc, int NH) {
  const int lane = threadIdx.x & 31;
  const int gw = (int)(((size_t)blockIdx.x * blockDim.x + threadIdx.x) >> 5);
  const int h = gw % NH;
  const int t = (gw / NH) % Tc;
  const int b = gw / (NH * Tc);
  if (b >= Bc) return;

  const float* src = X + (((size_t)b * Tc + t) * NH + h) * 64;
  float xr = src[2 * lane], xi = src[2 * lane + 1];
  float c = fcos[t * 32 + lane], s = fsin[t * 32 + lane];
  float orr = xr * c - xi * s;
  float oi  = xr * s + xi * c;

  float ss = orr * orr + oi * oi;
#pragma unroll
  for (int off = 1; off < 32; off <<= 1) ss += __shfl_xor(ss, off, 32);
  float sc = scales[h] / fmaxf(sqrtf(ss), 1e-12f);

  _Float16* dst = Out + (((size_t)b * NH + h) * Tc + t) * 64;
  dst[2 * lane]     = (_Float16)(orr * sc);
  dst[2 * lane + 1] = (_Float16)(oi * sc);
}

// [B,T,G,64] f32 -> [B,G,T,64] f16
__global__ void vtrans_kernel(const float* __restrict__ V, _Float16* __restrict__ Out,
                              int Bc, int Tc, int Gc) {
  size_t i = (size_t)blockIdx.x * blockDim.x + threadIdx.x;
  size_t total = (size_t)Bc * Tc * Gc * 64;
  if (i >= total) return;
  int d = (int)(i & 63);
  size_t r = i >> 6;
  int g = (int)(r % Gc);
  size_t r2 = r / Gc;
  int t = (int)(r2 % Tc);
  int b = (int)(r2 / Tc);
  Out[(((size_t)b * Gc + g) * Tc + t) * 64 + d] = (_Float16)V[i];
}

// ---------------------------------------------------------------------------
// Causal flash attention with GQA. Wave handles 16 query rows, block 128 rows.
// Q/K/V are [B,H|G,T,64] f16; output Y is [B,T,H*64] f16.
// V staged TRANSPOSED in LDS (row pitch 48 f16 = 96 B -> 16 B aligned b128
// gathers); P transposed through per-wave LDS with the same pitch so both
// A/B fragment gathers reuse load_frag (2 x ds_load_b128 each).
// K fragments and V fragments are batch-loaded ahead of their WMMA chains.
// ---------------------------------------------------------------------------
#define HD 64
__global__ void attn_kernel(const _Float16* __restrict__ Qh,
                            const _Float16* __restrict__ Kh,
                            const _Float16* __restrict__ Vh,
                            _Float16* __restrict__ Yh,
                            int Bc, int Hc, int Gc, int Tc) {
  __shared__ __align__(16) _Float16 vbufT[64][48];     // V^T chunk: [dim][key]
  __shared__ __align__(16) _Float16 pbuf[8][16][48];   // per-wave P scratch

  const int lane = threadIdx.x & 31;
  const int w    = threadIdx.x >> 5;
  const int half = lane >> 4;
  const int lid  = lane & 15;

  const int bh = blockIdx.x;
  const int h  = bh % Hc;
  const int b  = bh / Hc;
  const int g  = h / (Hc / Gc);
  const int q0blk = blockIdx.y * 128;
  const int q0 = q0blk + w * 16;

  const _Float16* Qp = Qh + ((size_t)(b * Hc + h) * Tc) * HD;
  const _Float16* Kp = Kh + ((size_t)(b * Gc + g) * Tc) * HD;
  const _Float16* Vp = Vh + ((size_t)(b * Gc + g) * Tc) * HD;

  // Q A-fragments for d = [0,32) and [32,64), resident for the whole loop.
  v16h qa0 = load_frag(Qp, HD, q0 + lid, 0, half);
  v16h qa1 = load_frag(Qp, HD, q0 + lid, 32, half);

  float mrow[8], lrow[8];
  v8f acc[4];
#pragma unroll
  for (int j = 0; j < 8; ++j) { mrow[j] = -3.0e38f; lrow[j] = 0.0f; }
#pragma unroll
  for (int nt = 0; nt < 4; ++nt)
#pragma unroll
    for (int j = 0; j < 8; ++j) acc[nt][j] = 0.0f;

  const float SCALE = 0.125f;  // 1/sqrt(64)
  int kend = q0blk + 128;
  if (kend > Tc) kend = Tc;

  for (int kc = 0; kc < kend; kc += 32) {
    __syncthreads();
    {  // cooperative transposed V stage: thread reads v8h row chunk, scatters
      int r = threadIdx.x >> 3, sg = threadIdx.x & 7;  // key row, dim segment
      v8h vrow = *(const v8h*)(Vp + (size_t)(kc + r) * HD + sg * 8);
#pragma unroll
      for (int i = 0; i < 8; ++i) vbufT[sg * 8 + i][r] = vrow[i];
    }
    __syncthreads();

    // batch all K fragments, then the 4 S WMMAs
    v16h kb00 = load_frag(Kp, HD, kc + lid,      0,  half);
    v16h kb01 = load_frag(Kp, HD, kc + lid,      32, half);
    v16h kb10 = load_frag(Kp, HD, kc + 16 + lid, 0,  half);
    v16h kb11 = load_frag(Kp, HD, kc + 16 + lid, 32, half);

    v8f s[2];
#pragma unroll
    for (int kt = 0; kt < 2; ++kt)
#pragma unroll
      for (int j = 0; j < 8; ++j) s[kt][j] = 0.0f;
    s[0] = __builtin_amdgcn_wmma_f32_16x16x32_f16(false, qa0, false, kb00,
                                                  (short)0, s[0], false, false);
    s[1] = __builtin_amdgcn_wmma_f32_16x16x32_f16(false, qa0, false, kb10,
                                                  (short)0, s[1], false, false);
    s[0] = __builtin_amdgcn_wmma_f32_16x16x32_f16(false, qa1, false, kb01,
                                                  (short)0, s[0], false, false);
    s[1] = __builtin_amdgcn_wmma_f32_16x16x32_f16(false, qa1, false, kb11,
                                                  (short)0, s[1], false, false);

    // scale + causal mask + online softmax (row spans 16 lanes -> xor reduce)
    float newm[8];
#pragma unroll
    for (int j = 0; j < 8; ++j) {
      int qrow = q0 + half * 8 + j;
      float v0 = s[0][j] * SCALE;
      float v1 = s[1][j] * SCALE;
      if (kc + lid > qrow)      v0 = -3.0e38f;
      if (kc + 16 + lid > qrow) v1 = -3.0e38f;
      s[0][j] = v0; s[1][j] = v1;
      float mx = fmaxf(v0, v1);
#pragma unroll
      for (int off = 1; off < 16; off <<= 1) mx = fmaxf(mx, __shfl_xor(mx, off, 32));
      newm[j] = fmaxf(mrow[j], mx);
    }
#pragma unroll
    for (int j = 0; j < 8; ++j) {
      float corr = __expf(mrow[j] - newm[j]);
      float e0 = __expf(s[0][j] - newm[j]);
      float e1 = __expf(s[1][j] - newm[j]);
      pbuf[w][half * 8 + j][lid]      = (_Float16)e0;  // C-layout -> LDS
      pbuf[w][half * 8 + j][16 + lid] = (_Float16)e1;
      float rs = e0 + e1;
#pragma unroll
      for (int off = 1; off < 16; off <<= 1) rs += __shfl_xor(rs, off, 32);
      lrow[j] = lrow[j] * corr + rs;
      mrow[j] = newm[j];
#pragma unroll
      for (int nt = 0; nt < 4; ++nt) acc[nt][j] *= corr;
    }

    // P fragment + all 4 V fragments first, then the 4 PV WMMAs
    v16h pa  = load_frag(&pbuf[w][0][0], 48, lid, 0, half);
    v16h vb0 = load_frag(&vbufT[0][0], 48,  0 + lid, 0, half);
    v16h vb1 = load_frag(&vbufT[0][0], 48, 16 + lid, 0, half);
    v16h vb2 = load_frag(&vbufT[0][0], 48, 32 + lid, 0, half);
    v16h vb3 = load_frag(&vbufT[0][0], 48, 48 + lid, 0, half);
    acc[0] = __builtin_amdgcn_wmma_f32_16x16x32_f16(false, pa, false, vb0,
                                                    (short)0, acc[0], false, false);
    acc[1] = __builtin_amdgcn_wmma_f32_16x16x32_f16(false, pa, false, vb1,
                                                    (short)0, acc[1], false, false);
    acc[2] = __builtin_amdgcn_wmma_f32_16x16x32_f16(false, pa, false, vb2,
                                                    (short)0, acc[2], false, false);
    acc[3] = __builtin_amdgcn_wmma_f32_16x16x32_f16(false, pa, false, vb3,
                                                    (short)0, acc[3], false, false);
  }

  // epilogue: divide by l, write back transposed to [B,T,H*64]
#pragma unroll
  for (int j = 0; j < 8; ++j) {
    float inv = 1.0f / fmaxf(lrow[j], 1e-20f);
    int t = q0 + half * 8 + j;
    _Float16* dst = Yh + ((size_t)b * Tc + t) * (size_t)(Hc * HD) + h * HD;
#pragma unroll
    for (int nt = 0; nt < 4; ++nt)
      dst[nt * 16 + lid] = (_Float16)(acc[nt][j] * inv);
  }
}

// ---------------------------------------------------------------------------
extern "C" void kernel_launch(void* const* d_in, const int* in_sizes, int n_in,
                              void* d_out, int out_size, void* d_ws, size_t ws_size,
                              hipStream_t stream) {
  (void)in_sizes; (void)n_in; (void)out_size; (void)ws_size;
  const float* x     = (const float*)d_in[0];
  const float* fcos  = (const float*)d_in[1];
  const float* fsin  = (const float*)d_in[2];
  const float* wq    = (const float*)d_in[3];
  const float* wk    = (const float*)d_in[4];
  const float* wv    = (const float*)d_in[5];
  const float* wo    = (const float*)d_in[6];
  const float* qscal = (const float*)d_in[7];
  const float* kscal = (const float*)d_in[8];
  float* out = (float*)d_out;

  constexpr int Bc = 2, Tc = 2048, DIMc = 2048, Hc = 32, Gc = 8, KV = 512;
  const size_t BT = (size_t)Bc * Tc;  // 4096

  char* p = (char*)d_ws;
  auto alloc = [&](size_t bytes) {
    char* r = p;
    p += (bytes + 255) & ~(size_t)255;
    return r;
  };
  _Float16* xh  = (_Float16*)alloc(BT * DIMc * 2);
  _Float16* wqh = (_Float16*)alloc((size_t)DIMc * DIMc * 2);
  _Float16* wkh = (_Float16*)alloc((size_t)KV * DIMc * 2);
  _Float16* wvh = (_Float16*)alloc((size_t)KV * DIMc * 2);
  _Float16* woh = (_Float16*)alloc((size_t)DIMc * DIMc * 2);
  float*    qf  = (float*)alloc(BT * DIMc * 4);
  float*    kf  = (float*)alloc(BT * KV * 4);
  float*    vf  = (float*)alloc(BT * KV * 4);
  _Float16* qh  = (_Float16*)alloc(BT * DIMc * 2);
  _Float16* kh  = (_Float16*)alloc(BT * KV * 2);
  _Float16* vh  = (_Float16*)alloc(BT * KV * 2);
  _Float16* yh  = (_Float16*)alloc(BT * DIMc * 2);

  auto cvt = [&](const float* src, _Float16* dst, size_t n) {
    cvt_kernel<<<dim3((unsigned)((n + 255) / 256)), 256, 0, stream>>>(src, dst, n);
  };
  cvt(x,  xh,  BT * DIMc);
  cvt(wq, wqh, (size_t)DIMc * DIMc);
  cvt(wk, wkh, (size_t)KV * DIMc);
  cvt(wv, wvh, (size_t)KV * DIMc);
  cvt(wo, woh, (size_t)DIMc * DIMc);

  // projections: q/k/v = x @ W^T
  gemm_f16_wmma<<<dim3(32, 32), 256, 0, stream>>>(xh, wqh, qf, 4096, 2048, 2048);
  gemm_f16_wmma<<<dim3(32, 8),  256, 0, stream>>>(xh, wkh, kf, 4096, 512,  2048);
  gemm_f16_wmma<<<dim3(32, 8),  256, 0, stream>>>(xh, wvh, vf, 4096, 512,  2048);

  // RoPE + l2norm + head-scale + transpose to [B,H|G,T,64] f16
  rope_norm_kernel<<<dim3((Bc * Tc * Hc) / 8), 256, 0, stream>>>(qf, fcos, fsin, qscal, qh, Bc, Tc, Hc);
  rope_norm_kernel<<<dim3((Bc * Tc * Gc) / 8), 256, 0, stream>>>(kf, fcos, fsin, kscal, kh, Bc, Tc, Gc);
  vtrans_kernel<<<dim3((unsigned)((BT * KV) / 256)), 256, 0, stream>>>(vf, vh, Bc, Tc, Gc);

  // causal flash attention -> yh [B,T,2048] f16
  attn_kernel<<<dim3(Bc * Hc, Tc / 128), 256, 0, stream>>>(qh, kh, vh, yh, Bc, Hc, Gc, Tc);

  // output projection -> f32 d_out
  gemm_f16_wmma<<<dim3(32, 32), 256, 0, stream>>>(yh, woh, out, 4096, 2048, 2048);
}
